// Dylan_MT_Net_11768210391372
// MI455X (gfx1250) — matrix operand order repaired and checked by reading, not compiled, not run on hardware
//
#include <hip/hip_runtime.h>

// ============================================================================
// Dylan_MT_Net forward for MI455X (gfx1250, wave32, WMMA).
// All channel-mixing convs / FC layers / attention-score GEMMs run on
// v_wmma_f32_16x16x32_f16 (f16 inputs, f32 accumulate) with LDS-staged tiles:
//  - 128-thread block = 4 waves computing a 32x32 output tile (2x2 subtiles)
//  - per 32-K chunk: cooperative float4 (b128) global loads -> f16 -> LDS
//  - DOUBLE-BUFFERED K pipeline: chunk k+1 staged while chunk k is in the
//    matrix pipe; one s_barrier per chunk, global loads in flight across it
//  - per-lane fragments are contiguous LDS runs (2x/1x 16B ds loads)
// Bias + batchnorm + residual + leaky are fused into the GEMM epilogue.
// ============================================================================

constexpr int N_    = 128;            // batch
constexpr int T_    = 64;
constexpr int V_    = 22;
constexpr int S_    = 3;
constexpr int C_    = 64;
constexpr int CO_   = 128;
constexpr int P_    = T_ * V_;        // 1408 spatial positions
constexpr int FLAT_ = C_ * (T_ / 2) * V_;   // 45056
constexpr int LDSW  = 40;             // LDS row stride in halfs (16B aligned,
                                      // +8 pad to spread banks)
constexpr int TSZ   = 32 * LDSW;      // one 32x32 tile in halfs

typedef long long i64;
typedef __attribute__((ext_vector_type(16))) _Float16 v16h;
typedef __attribute__((ext_vector_type(8)))  _Float16 v8h;
typedef __attribute__((ext_vector_type(8)))  float    v8f;

// ----------------------------------------------------------------------------
// One 16x16x32 WMMA step sourced from LDS tiles.
//   As[row][k] : 32 rows (M) x 32 k, row stride LDSW halfs
//   Bt[col][k] : 32 cols (N) x 32 k (B stored transposed), row stride LDSW
// Fragment layouts per CDNA5 ISA 7.12.2:
//   A lane&15 = row, lane>=16 -> K+8 inside each 16-K group: two contiguous
//     8-half runs at k = kaH and 16+kaH.
//   B lane&15 = col, lanes>=16 -> K+8 inside each 16-K group: two contiguous
//     8-half runs at k = kbH and 16+kbH.
//   C/D: VGPR r -> row r + 8*(lane>=16), col = lane&15.
// ----------------------------------------------------------------------------
__device__ __forceinline__ v8f wmma_lds_step(const _Float16* __restrict__ As,
                                             const _Float16* __restrict__ Bt,
                                             int lane, int mS, int nS, v8f acc) {
  const int arow = mS + (lane & 15);
  const int kaH  = (lane >> 4) * 8;
  const int bcol = nS + (lane & 15);
  const int kbH  = (lane >> 4) * 8;
  const v8h a0 = *(const v8h*)(As + arow * LDSW + kaH);
  const v8h a1 = *(const v8h*)(As + arow * LDSW + 16 + kaH);
  const v8h b0 = *(const v8h*)(Bt + bcol * LDSW + kbH);
  const v8h b1 = *(const v8h*)(Bt + bcol * LDSW + 16 + kbH);
  v16h a, b;
#pragma unroll
  for (int j = 0; j < 8; ++j) {
    a[j] = a0[j]; a[j + 8] = a1[j];
    b[j] = b0[j]; b[j + 8] = b1[j];
  }
  return __builtin_amdgcn_wmma_f32_16x16x32_f16(false, a, false, b, (short)0,
                                                acc, false, false);
}

// ----------------------------------------------------------------------------
// Strided batched GEMM:  O[b] = act( bn( A*B[b] + bias ) + R[b] )
// Requirements (hold for every call site):
//   A(m,k) = A[m*sAm + k]        (row-contiguous weights, sAm % 4 == 0)
//   B(k,n) = B[b*sBb + k*sBk + n] (column-contiguous, sBk % 4 == 0)
//   K % 32 == 0, Ncols % 32 == 0; M may be ragged.
// O(m,n) at O[b*sOb + m*sOm + n*sOn]; R uses O's m/n strides.
// act: 0 = none, 1 = leaky(alpha) (alpha=0 -> relu)
// ----------------------------------------------------------------------------
__global__ __launch_bounds__(128) void k_gemm_wmma(
    const float* __restrict__ A, int sAm,
    const float* __restrict__ B, int sBk, i64 sBb,
    float* __restrict__ O, int sOm, int sOn, i64 sOb,
    const float* __restrict__ bias,
    const float* __restrict__ bnS, const float* __restrict__ bnB,
    const float* __restrict__ R, i64 sRb,
    float alpha, int act, int M, int Ncols, int K) {
  __shared__ _Float16 As[2 * TSZ];
  __shared__ _Float16 Bt[2 * TSZ];
  const int tid = threadIdx.x;
  const int lane = tid & 31, wave = tid >> 5;
  const int m0 = blockIdx.y * 32, n0 = blockIdx.x * 32;
  const int mS = (wave >> 1) * 16, nS = (wave & 1) * 16;
  const float* Bb = B + (i64)blockIdx.z * sBb + n0;

  const int r   = tid >> 2;        // 0..31: staged row
  const int seg = (tid & 3) * 8;   // 0,8,16,24: 8-wide column segment
  const int am  = m0 + r;
  const bool aok = (am < M);
  const float* Arow = A + (i64)(aok ? am : (M - 1)) * sAm + seg;
  const float* Brow = Bb + seg;

  auto stage = [&](int k0, int p) {
    // ---- A (zero-pad ragged rows; loads stay unconditional) ----
    float4 f0 = *(const float4*)(Arow + k0);
    float4 f1 = *(const float4*)(Arow + k0 + 4);
    if (!aok) { f0.x = f0.y = f0.z = f0.w = 0.f; f1 = f0; }
    v8h ha;
    ha[0] = (_Float16)f0.x; ha[1] = (_Float16)f0.y;
    ha[2] = (_Float16)f0.z; ha[3] = (_Float16)f0.w;
    ha[4] = (_Float16)f1.x; ha[5] = (_Float16)f1.y;
    ha[6] = (_Float16)f1.z; ha[7] = (_Float16)f1.w;
    *(v8h*)(As + p * TSZ + r * LDSW + seg) = ha;
    // ---- B transposed: Bt[n][k] ----
    const float* pb = Brow + (i64)(k0 + r) * sBk;
    float4 g0 = *(const float4*)pb;
    float4 g1 = *(const float4*)(pb + 4);
    _Float16 hb[8];
    hb[0] = (_Float16)g0.x; hb[1] = (_Float16)g0.y;
    hb[2] = (_Float16)g0.z; hb[3] = (_Float16)g0.w;
    hb[4] = (_Float16)g1.x; hb[5] = (_Float16)g1.y;
    hb[6] = (_Float16)g1.z; hb[7] = (_Float16)g1.w;
#pragma unroll
    for (int j = 0; j < 8; ++j)
      Bt[p * TSZ + (seg + j) * LDSW + r] = hb[j];
  };

  v8f acc = {0.f, 0.f, 0.f, 0.f, 0.f, 0.f, 0.f, 0.f};
  const int nk = K >> 5;
  stage(0, 0);
  __syncthreads();
  for (int kc = 0; kc < nk; ++kc) {
    const int cur = kc & 1;
    if (kc + 1 < nk) stage((kc + 1) << 5, cur ^ 1);   // prefetch next chunk
    acc = wmma_lds_step(As + cur * TSZ, Bt + cur * TSZ, lane, mS, nS, acc);
    __syncthreads();
  }

  const int nn = n0 + nS + (lane & 15);
  const int mb = m0 + mS + ((lane >> 4) << 3);
  if (nn < Ncols) {
    float* Ob = O + (i64)blockIdx.z * sOb;
    const float* Rb = R ? (R + (i64)blockIdx.z * sRb) : nullptr;
#pragma unroll
    for (int rr = 0; rr < 8; ++rr) {
      const int mm = mb + rr;
      if (mm >= M) continue;
      float v = acc[rr];
      if (bias) v += bias[mm];
      if (bnS)  v = v * bnS[mm] + bnB[mm];
      if (Rb)   v += Rb[(i64)mm * sOm + (i64)nn * sOn];
      if (act)  v = (v >= 0.f) ? v : alpha * v;
      Ob[(i64)mm * sOm + (i64)nn * sOn] = v;
    }
  }
}

// Spatial attention scores: per (n,s), G[u,v] = sum_{c,t} q[c,t,u]*k[c,t,v].
// Single 32x32 tile (22x22 valid), 4 waves, K = C*T = 4096.  grid = N*S.
__global__ __launch_bounds__(128) void k_att_scores_s(
    const float* __restrict__ qk, float* __restrict__ sc) {
  __shared__ _Float16 As[2 * TSZ];   // As[u][k]
  __shared__ _Float16 Bt[2 * TSZ];   // Bt[v][k]
  const int tid = threadIdx.x;
  const int lane = tid & 31, wave = tid >> 5;
  const int ns = blockIdx.x;
  const int n = ns / S_, s = ns % S_;
  const float* Q  = qk + ((i64)n * (2 * S_ * C_) + s * C_) * P_;
  const float* Kd = Q + (i64)(S_ * C_) * P_;
  const int mS = (wave >> 1) * 16, nS = (wave & 1) * 16;

  auto stage = [&](int k0, int p) {
#pragma unroll
    for (int e = 0; e < 8; ++e) {
      const int idx = e * 128 + tid;       // 0..1023
      const int kk = idx >> 5, uu = idx & 31;
      const int uc = (uu < V_) ? uu : 0;   // clamp; loads unconditional
      float qa = Q[(i64)(k0 + kk) * V_ + uc];
      float kb = Kd[(i64)(k0 + kk) * V_ + uc];
      if (uu >= V_) { qa = 0.f; kb = 0.f; }
      As[p * TSZ + uu * LDSW + kk] = (_Float16)qa;
      Bt[p * TSZ + uu * LDSW + kk] = (_Float16)kb;
    }
  };

  v8f acc = {0.f, 0.f, 0.f, 0.f, 0.f, 0.f, 0.f, 0.f};
  const int nk = (C_ * T_) >> 5;
  stage(0, 0);
  __syncthreads();
  for (int kc = 0; kc < nk; ++kc) {
    const int cur = kc & 1;
    if (kc + 1 < nk) stage((kc + 1) << 5, cur ^ 1);
    acc = wmma_lds_step(As + cur * TSZ, Bt + cur * TSZ, lane, mS, nS, acc);
    __syncthreads();
  }

  const int vv = nS + (lane & 15);
  const int mb = mS + ((lane >> 4) << 3);
  if (vv < V_) {
    float* out = sc + (i64)ns * V_ * V_;
#pragma unroll
    for (int rr = 0; rr < 8; ++rr) {
      const int uu = mb + rr;
      if (uu < V_) out[uu * V_ + vv] = acc[rr];
    }
  }
}

// Temporal attention scores: per (n,s), G[t,q] = sum_{c,v} q[c,t,v]*k[c,q,v].
// 64x64 output = 2x2 blocks of 32x32; K = C*V = 1408, k = c*V + v.
// grid = (4, N*S).
__global__ __launch_bounds__(128) void k_att_scores_t(
    const float* __restrict__ qk, float* __restrict__ sc) {
  __shared__ _Float16 As[2 * TSZ];   // As[t][k]
  __shared__ _Float16 Bt[2 * TSZ];   // Bt[q][k]
  const int tid = threadIdx.x;
  const int lane = tid & 31, wave = tid >> 5;
  const int ns = blockIdx.y;
  const int n = ns / S_, s = ns % S_;
  const int t0 = (blockIdx.x >> 1) * 32, q0 = (blockIdx.x & 1) * 32;
  const float* Q  = qk + ((i64)n * (2 * S_ * C_) + s * C_) * P_;
  const float* Kd = Q + (i64)(S_ * C_) * P_;
  const int mS = (wave >> 1) * 16, nS = (wave & 1) * 16;

  auto stage = [&](int k0, int p) {
#pragma unroll
    for (int e = 0; e < 8; ++e) {
      const int idx = e * 128 + tid;
      const int rr = idx >> 5, kk = idx & 31;
      const int k = k0 + kk;
      const int c = k / V_, v = k - c * V_;
      As[p * TSZ + rr * LDSW + kk] =
          (_Float16)Q[(i64)c * P_ + (t0 + rr) * V_ + v];
      Bt[p * TSZ + rr * LDSW + kk] =
          (_Float16)Kd[(i64)c * P_ + (q0 + rr) * V_ + v];
    }
  };

  v8f acc = {0.f, 0.f, 0.f, 0.f, 0.f, 0.f, 0.f, 0.f};
  const int nk = (C_ * V_) >> 5;
  stage(0, 0);
  __syncthreads();
  for (int kc = 0; kc < nk; ++kc) {
    const int cur = kc & 1;
    if (kc + 1 < nk) stage((kc + 1) << 5, cur ^ 1);
    acc = wmma_lds_step(As + cur * TSZ, Bt + cur * TSZ, lane, mS, nS, acc);
    __syncthreads();
  }

  const int qq = q0 + nS + (lane & 15);
  const int mb = t0 + mS + ((lane >> 4) << 3);
  float* out = sc + (i64)ns * T_ * T_;
#pragma unroll
  for (int rr = 0; rr < 8; ++rr) out[(mb + rr) * T_ + qq] = acc[rr];
}

// Row softmax (L<=64) with pre-scale and post-multiplier alpha[s],
// s = (row / perS) % S.  One wave per row.
__global__ __launch_bounds__(32) void k_softmax(
    float* __restrict__ d, const float* __restrict__ alpha, float scale, int L,
    int perS) {
  const int r = blockIdx.x;
  const int lane = threadIdx.x;
  float* row = d + (i64)r * L;
  const int s = (r / perS) % S_;
  float x0 = (lane < L) ? row[lane] * scale : -3.0e38f;
  float x1 = (lane + 32 < L) ? row[lane + 32] * scale : -3.0e38f;
  float m = fmaxf(x0, x1);
  for (int off = 16; off; off >>= 1) m = fmaxf(m, __shfl_xor(m, off, 32));
  float e0 = (lane < L) ? __expf(x0 - m) : 0.f;
  float e1 = (lane + 32 < L) ? __expf(x1 - m) : 0.f;
  float t = e0 + e1;
  for (int off = 16; off; off >>= 1) t += __shfl_xor(t, off, 32);
  const float a = alpha[s] / t;
  if (lane < L) row[lane] = e0 * a;
  if (lane + 32 < L) row[lane + 32] = e1 * a;
}

// y[n, s*C+c, t, v] = sum_u x[n,c,t,u] * att[n,s,u,v].  Thread per (n,s,c,t).
__global__ void k_apply_att_s(const float* __restrict__ x,
                              const float* __restrict__ att,
                              float* __restrict__ y) {
  const i64 id = (i64)blockIdx.x * blockDim.x + threadIdx.x;
  if (id >= (i64)N_ * S_ * C_ * T_) return;
  const int t = (int)(id & 63);
  const int c = (int)((id >> 6) & 63);
  const int s = (int)((id >> 12) % S_);
  const int n = (int)(id / ((i64)S_ * C_ * T_));
  const float* xr = x + ((i64)n * C_ + c) * P_ + (i64)t * V_;
  const float* at = att + (i64)(n * S_ + s) * V_ * V_;
  float o[V_];
#pragma unroll
  for (int v = 0; v < V_; ++v) o[v] = 0.f;
  for (int u = 0; u < V_; ++u) {
    const float xv = xr[u];
    const float* ar = at + u * V_;
#pragma unroll
    for (int v = 0; v < V_; ++v) o[v] += xv * ar[v];
  }
  float* yr = y + ((i64)n * (S_ * C_) + s * C_ + c) * P_ + (i64)t * V_;
#pragma unroll
  for (int v = 0; v < V_; ++v) yr[v] = o[v];
}

// z[n, s*C+c, q, v] = sum_t yin[n,c,t,v] * attt[n,s,t,q]. Thread per (n,s,c,q).
__global__ void k_apply_att_t(const float* __restrict__ yin,
                              const float* __restrict__ att,
                              float* __restrict__ z) {
  const i64 id = (i64)blockIdx.x * blockDim.x + threadIdx.x;
  if (id >= (i64)N_ * S_ * C_ * T_) return;
  const int q = (int)(id & 63);
  const int c = (int)((id >> 6) & 63);
  const int s = (int)((id >> 12) % S_);
  const int n = (int)(id / ((i64)S_ * C_ * T_));
  const float* yr = yin + ((i64)n * C_ + c) * P_;
  const float* at = att + (i64)(n * S_ + s) * T_ * T_;
  float o[V_];
#pragma unroll
  for (int v = 0; v < V_; ++v) o[v] = 0.f;
  for (int t = 0; t < T_; ++t) {
    const float a = at[(i64)t * T_ + q];
    const float* yv = yr + (i64)t * V_;
#pragma unroll
    for (int v = 0; v < V_; ++v) o[v] += yv[v] * a;
  }
  float* zr = z + ((i64)n * (S_ * C_) + s * C_ + c) * P_ + (i64)q * V_;
#pragma unroll
  for (int v = 0; v < V_; ++v) zr[v] = o[v];
}

// 3x3 SAME conv over (T,V), 3 in-ch, 64 out-ch, + bias; writes hs = h + pe_s
// and ht = hs + pe_t.  x layout (N,T,V,3); w OIHW (64,3,3,3).
__global__ void k_proj(const float* __restrict__ x, const float* __restrict__ w,
                       const float* __restrict__ b, const float* __restrict__ pes,
                       const float* __restrict__ pet, float* __restrict__ hs,
                       float* __restrict__ ht) {
  const i64 id = (i64)blockIdx.x * blockDim.x + threadIdx.x;
  if (id >= (i64)N_ * C_ * P_) return;
  const int v = (int)(id % V_);
  const int t = (int)((id / V_) % T_);
  const int o = (int)((id / P_) % C_);
  const int n = (int)(id / ((i64)C_ * P_));
  float acc = b[o];
  for (int dt = -1; dt <= 1; ++dt) {
    const int tt = t + dt;
    if (tt < 0 || tt >= T_) continue;
    for (int dv = -1; dv <= 1; ++dv) {
      const int vv = v + dv;
      if (vv < 0 || vv >= V_) continue;
      const float* xp = x + (((i64)n * T_ + tt) * V_ + vv) * 3;
      const float* wp = w + (i64)o * 27 + (dt + 1) * 3 + (dv + 1);
      for (int i = 0; i < 3; ++i) acc += xp[i] * wp[i * 9];
    }
  }
  const i64 idx = ((i64)n * C_ + o) * P_ + (i64)t * V_ + v;
  const i64 pidx = ((i64)o * T_ + t) * V_ + v;
  const float h = acc + pes[pidx];
  hs[idx] = h;
  ht[idx] = h + pet[pidx];
}

__global__ void k_add(const float* __restrict__ a, const float* __restrict__ b,
                      float* __restrict__ o, i64 total) {
  const i64 id = (i64)blockIdx.x * blockDim.x + threadIdx.x;
  if (id < total) o[id] = a[id] + b[id];
}

// mean over (t,v); output transposed (C x N) so following GEMMs are row-major.
__global__ __launch_bounds__(32) void k_mean_cn(const float* __restrict__ x,
                                                float* __restrict__ mT) {
  const int nc = blockIdx.x;          // n*C + c
  const int n = nc >> 6, c = nc & 63;
  const int lane = threadIdx.x;
  const float* p = x + (i64)nc * P_;
  float s = 0.f;
  for (int i = lane; i < P_; i += 32) s += p[i];
  for (int off = 16; off; off >>= 1) s += __shfl_xor(s, off, 32);
  if (lane == 0) mT[(i64)c * N_ + n] = s * (1.f / P_);
}

// wei = sigmoid(xl + xg); o = x*wei + res*(1-wei).  xg is (C x N).
__global__ void k_sigmix(const float* __restrict__ x, const float* __restrict__ res,
                         const float* __restrict__ xl, const float* __restrict__ xg,
                         float* __restrict__ o) {
  const i64 id = (i64)blockIdx.x * blockDim.x + threadIdx.x;
  if (id >= (i64)N_ * C_ * P_) return;
  const int n = (int)(id / ((i64)C_ * P_));
  const int c = (int)((id / P_) % C_);
  const float t = xl[id] + xg[(i64)c * N_ + n];
  const float w = 1.f / (1.f + __expf(-t));
  o[id] = x[id] * w + res[id] * (1.f - w);
}

// max-pool over time pairs, write flattened transposed (FLAT x N).
__global__ void k_pool(const float* __restrict__ z, float* __restrict__ zT) {
  const i64 id = (i64)blockIdx.x * blockDim.x + threadIdx.x;
  if (id >= (i64)N_ * FLAT_) return;
  const int v = (int)(id % V_);
  const int th = (int)((id / V_) % (T_ / 2));
  const int c = (int)((id / (V_ * (T_ / 2))) % C_);
  const int n = (int)(id / FLAT_);
  const float* p = z + ((i64)n * C_ + c) * P_ + (i64)(2 * th) * V_ + v;
  const i64 f = ((i64)c * (T_ / 2) + th) * V_ + v;
  zT[f * N_ + n] = fmaxf(p[0], p[V_]);
}

// ---------------------------------------------------------------------------
// Host side
// ---------------------------------------------------------------------------
static void launch_gemm(hipStream_t st, const float* A, int sAm,
                        const float* B, int sBk, i64 sBb, float* O, int sOm,
                        int sOn, i64 sOb, const float* bias, const float* bnS,
                        const float* bnB, const float* R, i64 sRb, float alpha,
                        int act, int M, int Ncols, int K, int batch) {
  dim3 g((unsigned)(Ncols / 32), (unsigned)((M + 31) / 32), (unsigned)batch);
  k_gemm_wmma<<<g, dim3(128), 0, st>>>(A, sAm, B, sBk, sBb, O, sOm, sOn, sOb,
                                       bias, bnS, bnB, R, sRb, alpha, act, M,
                                       Ncols, K);
}

extern "C" void kernel_launch(void* const* d_in, const int* in_sizes, int n_in,
                              void* d_out, int out_size, void* d_ws,
                              size_t ws_size, hipStream_t stream) {
  (void)in_sizes; (void)n_in; (void)out_size; (void)ws_size;
  auto F = [&](int i) { return (const float*)d_in[i]; };

  // Input flattening assumed: setup_inputs() insertion order, depth-first
  // (x, proj_w, proj_b, pe_s, pe_t, blocks[0..1] (22 leaves each: bn tuples
  // flatten as scale,bias), fusion (loc,glob,loc2 x 8), l1..l3, fc).
  const float* Xin = F(0);
  const float* proj_w = F(1); const float* proj_b = F(2);
  const float* pe_s = F(3);   const float* pe_t = F(4);
  struct Blk {
    const float *inw,*inb,*alps,*ow,*ob,*obnS,*obnB,*fw,*fb,*fbnS,*fbnB;
    const float *inwT,*inbT,*alpt,*owT,*obT,*obnTS,*obnTB,*fwT,*fbT,*fbnTS,*fbnTB;
  } bk[2];
  for (int i = 0; i < 2; ++i) {
    const int b = 5 + 22 * i;
    bk[i] = {F(b+0),F(b+1),F(b+2),F(b+3),F(b+4),F(b+5),F(b+6),F(b+7),F(b+8),
             F(b+9),F(b+10),F(b+11),F(b+12),F(b+13),F(b+14),F(b+15),F(b+16),
             F(b+17),F(b+18),F(b+19),F(b+20),F(b+21)};
  }
  struct AP { const float *w1,*b1,*bn1S,*bn1B,*w2,*b2,*bn2S,*bn2B; };
  auto getAP = [&](int b) {
    return AP{F(b),F(b+1),F(b+2),F(b+3),F(b+4),F(b+5),F(b+6),F(b+7)};
  };
  AP loc = getAP(49), glob = getAP(57), loc2 = getAP(65);
  const float *l1w=F(73),*l1b=F(74),*l1s=F(75),*l1bb=F(76);
  const float *l2w=F(77),*l2b=F(78),*l2s=F(79),*l2bb=F(80);
  const float *l3w=F(81),*l3b=F(82),*l3s=F(83),*l3bb=F(84);
  const float *fcw=F(85),*fcb=F(86);

  // ---- workspace layout (floats) ----
  float* ws = (float*)d_ws;
  const i64 NCP = (i64)N_ * C_ * P_;               // 11,534,336
  float* buf[5];
  for (int i = 0; i < 5; ++i) buf[i] = ws + (i64)i * NCP;
  float* qk    = ws + 5 * NCP;                     // N*384*P = 69,206,016
  float* ybuf  = qk;                               // alias: never live together
  float* cobuf = qk;                               // alias: fusion CO buffer
  float* sc    = qk + (i64)N_ * 2 * S_ * C_ * P_;  // max N*S*T*T
  float* mT    = sc + (i64)N_ * S_ * T_ * T_;      // (C x N)
  float* g1    = mT + (i64)C_ * N_;                // (CO x N)
  float* xg    = g1 + (i64)CO_ * N_;               // (C x N)
  float* zT    = xg + (i64)C_ * N_;                // (FLAT x N)
  float* fc1   = zT + (i64)FLAT_ * N_;
  float* fc2   = fc1 + (i64)512 * N_;
  float* fc3   = fc2 + (i64)256 * N_;

  // ---- stem: 3x3 conv + positional encodings ----
  {
    const i64 tot = NCP;
    k_proj<<<(unsigned)((tot + 255) / 256), 256, 0, stream>>>(
        Xin, proj_w, proj_b, pe_s, pe_t, buf[0], buf[1]);
  }
  float* xx = buf[0];
  float* yy = buf[1];
  float* fl[3] = {buf[2], buf[3], buf[4]};

  const i64 totNCP = NCP;
  const i64 totA = (i64)N_ * S_ * C_ * T_;

  // ---- 2 ST blocks ----
  for (int i = 0; i < 2; ++i) {
    const Blk& B = bk[i];
    float* t1 = fl[0]; float* xxn = fl[1]; float* yyn = fl[2];
    // spatial stream
    launch_gemm(stream, B.inw, C_, xx, P_, (i64)C_*P_, qk, P_, 1,
                (i64)2*S_*C_*P_, B.inb, nullptr, nullptr, nullptr, 0, 0.f, 0,
                2*S_*C_, P_, C_, N_);
    k_att_scores_s<<<dim3(N_ * S_), 128, 0, stream>>>(qk, sc);
    k_softmax<<<dim3(N_ * S_ * V_), 32, 0, stream>>>(sc, B.alps,
                                                     1.f / (C_ * T_), V_, V_);
    k_apply_att_s<<<(unsigned)((totA + 255) / 256), 256, 0, stream>>>(xx, sc,
                                                                      ybuf);
    launch_gemm(stream, B.ow, S_*C_, ybuf, P_, (i64)S_*C_*P_, t1, P_, 1,
                (i64)C_*P_, B.ob, B.obnS, B.obnB, xx, (i64)C_*P_, 0.1f, 1, C_,
                P_, S_*C_, N_);
    launch_gemm(stream, B.fw, C_, t1, P_, (i64)C_*P_, xxn, P_, 1,
                (i64)C_*P_, B.fb, B.fbnS, B.fbnB, xx, (i64)C_*P_, 0.1f, 1, C_,
                P_, C_, N_);
    // temporal stream (residuals use the OLD xx = block input)
    launch_gemm(stream, B.inwT, C_, yy, P_, (i64)C_*P_, qk, P_, 1,
                (i64)2*S_*C_*P_, B.inbT, nullptr, nullptr, nullptr, 0, 0.f, 0,
                2*S_*C_, P_, C_, N_);
    k_att_scores_t<<<dim3(4, N_ * S_), 128, 0, stream>>>(qk, sc);
    k_softmax<<<dim3(N_ * S_ * T_), 32, 0, stream>>>(sc, B.alpt,
                                                     1.f / (C_ * V_), T_, T_);
    k_apply_att_t<<<(unsigned)((totA + 255) / 256), 256, 0, stream>>>(yy, sc,
                                                                      ybuf);
    float* t3 = t1;  // t1 dead after ff
    launch_gemm(stream, B.owT, S_*C_, ybuf, P_, (i64)S_*C_*P_, t3, P_, 1,
                (i64)C_*P_, B.obT, B.obnTS, B.obnTB, xx, (i64)C_*P_, 0.1f, 1,
                C_, P_, S_*C_, N_);
    launch_gemm(stream, B.fwT, C_, t3, P_, (i64)C_*P_, yyn, P_, 1,
                (i64)C_*P_, B.fbT, B.fbnTS, B.fbnTB, xx, (i64)C_*P_, 0.1f, 1,
                C_, P_, C_, N_);
    float* o0 = xx; float* o1 = yy; float* o2 = t1;
    xx = xxn; yy = yyn;
    fl[0] = o0; fl[1] = o1; fl[2] = o2;
  }

  // ---- iAFF fusion ----
  float* xa = fl[0]; float* xl = fl[1]; float* xi = fl[2];
  k_add<<<(unsigned)((totNCP + 255) / 256), 256, 0, stream>>>(xx, yy, xa,
                                                              totNCP);
  auto att_path = [&](const AP& p, const float* in, float* out) {
    launch_gemm(stream, p.w1, C_, in, P_, (i64)C_*P_, cobuf, P_, 1,
                (i64)CO_*P_, p.b1, p.bn1S, p.bn1B, nullptr, 0, 0.f, 1, CO_, P_,
                C_, N_);                                    // relu (alpha=0)
    launch_gemm(stream, p.w2, CO_, cobuf, P_, (i64)CO_*P_, out, P_, 1,
                (i64)C_*P_, p.b2, p.bn2S, p.bn2B, nullptr, 0, 0.f, 0, C_, P_,
                CO_, N_);
  };
  auto glob_path = [&](const float* in) {   // in: (N,C,P) -> xg (C x N)
    k_mean_cn<<<dim3(N_ * C_), 32, 0, stream>>>(in, mT);
    launch_gemm(stream, glob.w1, C_, mT, N_, 0, g1, N_, 1, 0, glob.b1,
                glob.bn1S, glob.bn1B, nullptr, 0, 0.f, 1, CO_, N_, C_, 1);
    launch_gemm(stream, glob.w2, CO_, g1, N_, 0, xg, N_, 1, 0, glob.b2,
                glob.bn2S, glob.bn2B, nullptr, 0, 0.f, 0, C_, N_, CO_, 1);
  };
  att_path(loc, xa, xl);
  glob_path(xa);
  k_sigmix<<<(unsigned)((totNCP + 255) / 256), 256, 0, stream>>>(xx, yy, xl,
                                                                 xg, xi);
  att_path(loc2, xi, xl);
  glob_path(xi);
  float* zf = xa;  // xa dead
  k_sigmix<<<(unsigned)((totNCP + 255) / 256), 256, 0, stream>>>(xx, yy, xl,
                                                                 xg, zf);

  // ---- pool + FC head ----
  {
    const i64 tot = (i64)N_ * FLAT_;
    k_pool<<<(unsigned)((tot + 255) / 256), 256, 0, stream>>>(zf, zT);
  }
  launch_gemm(stream, l1w, FLAT_, zT, N_, 0, fc1, N_, 1, 0, l1b, l1s, l1bb,
              nullptr, 0, 0.2f, 1, 512, N_, FLAT_, 1);
  launch_gemm(stream, l2w, 512, fc1, N_, 0, fc2, N_, 1, 0, l2b, l2s, l2bb,
              nullptr, 0, 0.2f, 1, 256, N_, 512, 1);
  launch_gemm(stream, l3w, 256, fc2, N_, 0, fc3, N_, 1, 0, l3b, l3s, l3bb,
              nullptr, 0, 0.2f, 1, 256, N_, 256, 1);
  // final: out[n,class] -> sOm=1 (class stride), sOn=28 (batch stride)
  launch_gemm(stream, fcw, 256, fc3, N_, 0, (float*)d_out, 1, 28, 0, fcb,
              nullptr, nullptr, nullptr, 0, 0.f, 0, 28, N_, 256, 1);
}
